// XCAC_15564961480780
// MI455X (gfx1250) — matrix operand order repaired and checked by reading, not compiled, not run on hardware
//
#include <hip/hip_runtime.h>
#include <hip/hip_bf16.h>
#include <math.h>

#define DEV __device__ __forceinline__

typedef __bf16 bf16_t;
typedef __attribute__((ext_vector_type(16))) __bf16 v16bf;
typedef __attribute__((ext_vector_type(8)))  __bf16 v8bf;
typedef __attribute__((ext_vector_type(8)))  float  v8f;

// Problem constants
static constexpr int Bn  = 8;
static constexpr int Nn  = 4096;
static constexpr int Cn  = 768;
static constexpr int Hn  = 12;
static constexpr int DHn = 64;
static constexpr int Mtot = Bn * Nn;    // 32768
static constexpr int C3   = 3 * Cn;     // 2304

// ---------------------------------------------------------------------------
// WMMA fragment loaders (layouts per CDNA5 ISA 7.12.2, bf16 16x16x32)
// A matrix 16x32 (MxK): lane holds row M=lane%16; K chunks of 8 at
//   koff = (lane/16)*8 and 16+(lane/16)*8.
// B matrix 32x16 (KxN): lane holds col N=lane%16; K chunks of 8 at
//   koff = (lane/16)*8 and 16+(lane/16)*8.
// C/D 16x16 f32: elem v -> row = v + 8*(lane/16), col = lane%16.
// ---------------------------------------------------------------------------
DEV v16bf load_a_frag(const bf16_t* __restrict__ base, int ld) {
  int lane = threadIdx.x & 31;
  const bf16_t* p = base + (size_t)(lane & 15) * ld + ((lane >> 4) << 3);
  v8bf lo = *(const v8bf*)(p);
  v8bf hi = *(const v8bf*)(p + 16);
  return __builtin_shufflevector(lo, hi, 0,1,2,3,4,5,6,7,8,9,10,11,12,13,14,15);
}

DEV v16bf load_b_frag(const bf16_t* __restrict__ base, int ld) {
  int lane = threadIdx.x & 31;
  const bf16_t* p = base + (size_t)(lane & 15) * ld + ((lane >> 4) << 3);
  v8bf lo = *(const v8bf*)(p);
  v8bf hi = *(const v8bf*)(p + 16);
  return __builtin_shufflevector(lo, hi, 0,1,2,3,4,5,6,7,8,9,10,11,12,13,14,15);
}

DEV v8f wmma_bf16(v16bf a, v16bf b, v8f c) {
  return __builtin_amdgcn_wmma_f32_16x16x32_bf16(
      /*neg_a=*/false, a, /*neg_b=*/false, b,
      /*c_mod=*/(short)0, c, /*reuse_a=*/false, /*reuse_b=*/false);
}

DEV void pf(const void* p) { __builtin_prefetch(p, 0, 3); }

DEV unsigned pack2(float a, float b) {
  union { bf16_t h[2]; unsigned u; } pk;
  pk.h[0] = (bf16_t)a;
  pk.h[1] = (bf16_t)b;
  return pk.u;
}

// ---------------------------------------------------------------------------
// Kernel 0: f32 -> bf16 convert, 8 elements/thread (b128 in/out)
// n must be a multiple of 8 (all our tensors are).
// ---------------------------------------------------------------------------
__global__ void k_convert_bf16(const float* __restrict__ src,
                               bf16_t* __restrict__ dst, int n8) {
  for (int i = blockIdx.x * blockDim.x + threadIdx.x; i < n8;
       i += gridDim.x * blockDim.x) {
    int i8 = i * 8;
    v8f xv = *(const v8f*)(src + i8);
    v8bf o;
#pragma unroll
    for (int t = 0; t < 8; ++t) o[t] = (bf16_t)xv[t];
    *(v8bf*)(dst + i8) = o;
  }
}

// ---------------------------------------------------------------------------
// Kernel 1: QKV GEMM.  y[m, c] = sum_k x[m,k] * w_qkv[c,k]
//   m = b*4096 + n   (M = 32768),  c in [0, 2304),  K = 768
// Wave tile 32x64 (2 Mtiles x 4 Ntiles); block = 8 waves -> 64 x 256.
// Epilogue scatters:
//   which==0 -> q_s[(bh*64+d)*4096 + n]   (bf16, [B,H,DH,N], packed b32 stores)
//   which==1 -> k_s  same layout
//   which==2 -> v_s[(bh*4096+n)*64 + d]   (bf16, [B,H,N,DH])
// ---------------------------------------------------------------------------
__global__ void k_gemm_qkv(const bf16_t* __restrict__ xb,
                           const bf16_t* __restrict__ wb,
                           bf16_t* __restrict__ q_s,
                           bf16_t* __restrict__ k_s,
                           bf16_t* __restrict__ v_s) {
  int w    = threadIdx.x >> 5;
  int lane = threadIdx.x & 31;
  int wm = w & 1, wn = w >> 1;
  int m0 = blockIdx.y * 64 + wm * 32;
  int n0 = blockIdx.x * 256 + wn * 64;

  v8f acc[2][4];
#pragma unroll
  for (int i = 0; i < 2; ++i)
#pragma unroll
    for (int j = 0; j < 4; ++j) acc[i][j] = (v8f)(0.0f);

  for (int k = 0; k < Cn; k += 32) {
    // prefetch the streaming A operand two K-tiles ahead
    pf(xb + (size_t)m0 * Cn + k + 64);
    pf(xb + (size_t)(m0 + 16) * Cn + k + 64);
    v16bf a0 = load_a_frag(xb + (size_t)m0 * Cn + k, Cn);
    v16bf a1 = load_a_frag(xb + (size_t)(m0 + 16) * Cn + k, Cn);
#pragma unroll
    for (int j = 0; j < 4; ++j) {
      v16bf bf = load_b_frag(wb + (size_t)(n0 + j * 16) * Cn + k, Cn);
      acc[0][j] = wmma_bf16(a0, bf, acc[0][j]);
      acc[1][j] = wmma_bf16(a1, bf, acc[1][j]);
    }
  }

  int bq = m0 >> 12;              // batch index (uniform: 64-row blocks never
                                  // cross the 4096-row batch boundary)
#pragma unroll
  for (int j = 0; j < 4; ++j) {
    int c = n0 + j * 16 + (lane & 15);
    int which = c / Cn;           // wave-uniform (tiles 16-aligned, 768%16==0)
    int r = c % Cn;
    int h = r >> 6;
    int d = r & 63;
    int bh = bq * Hn + h;
#pragma unroll
    for (int i = 0; i < 2; ++i) {
      int mrow = m0 + i * 16 + ((lane >> 4) << 3);
      int n = mrow & 4095;
      if (which == 2) {
        bf16_t* vp = v_s + ((size_t)bh * Nn + n) * 64 + d;
#pragma unroll
        for (int v = 0; v < 8; ++v) vp[(size_t)v * 64] = (bf16_t)acc[i][j][v];
      } else {
        bf16_t* dst = (which == 0 ? q_s : k_s) + (size_t)(bh * 64 + d) * Nn + n;
        unsigned* du = (unsigned*)dst;   // n is 8-aligned -> 4B aligned
#pragma unroll
        for (int t = 0; t < 4; ++t)
          du[t] = pack2(acc[i][j][2 * t], acc[i][j][2 * t + 1]);
      }
    }
  }
}

// ---------------------------------------------------------------------------
// Kernel 2: row L2-norm reciprocals. One block per (bh,d) row (6144 rows).
// inv = 1 / max(sqrt(sum x^2), 1e-12)
// ---------------------------------------------------------------------------
__global__ void k_norms(const bf16_t* __restrict__ q_s,
                        const bf16_t* __restrict__ k_s,
                        float* __restrict__ invq, float* __restrict__ invk) {
  int row = blockIdx.x;
  const bf16_t* qp = q_s + (size_t)row * Nn;
  const bf16_t* kp = k_s + (size_t)row * Nn;
  float sq = 0.f, sk = 0.f;
  for (int i = threadIdx.x * 8; i < Nn; i += 256 * 8) {
    v8bf a = *(const v8bf*)(qp + i);
    v8bf b = *(const v8bf*)(kp + i);
#pragma unroll
    for (int t = 0; t < 8; ++t) {
      float fa = (float)a[t];
      float fb = (float)b[t];
      sq += fa * fa;
      sk += fb * fb;
    }
  }
  __shared__ float s1[256], s2[256];
  s1[threadIdx.x] = sq;
  s2[threadIdx.x] = sk;
  __syncthreads();
  for (int off = 128; off > 0; off >>= 1) {
    if (threadIdx.x < off) {
      s1[threadIdx.x] += s1[threadIdx.x + off];
      s2[threadIdx.x] += s2[threadIdx.x + off];
    }
    __syncthreads();
  }
  if (threadIdx.x == 0) {
    invq[row] = 1.0f / fmaxf(sqrtf(s1[0]), 1e-12f);
    invk[row] = 1.0f / fmaxf(sqrtf(s2[0]), 1e-12f);
  }
}

// ---------------------------------------------------------------------------
// Kernel 3: per-(b,h) gram matrix S[d,e] = sum_n q[d,n]k[e,n] (WMMA, K=4096),
// then scale by invq[d]*invk[e]*temperature[h], + tem, row softmax,
// store attn bf16 row-major [d][e] (e-contiguous -> B-operand layout).
// Block = 8 waves; wave w: tile row tm=w%4, tile-col pair tg=w/4.
// ---------------------------------------------------------------------------
__global__ void k_attn(const bf16_t* __restrict__ q_s,
                       const bf16_t* __restrict__ k_s,
                       const float* __restrict__ invq,
                       const float* __restrict__ invk,
                       const float* __restrict__ temperature,
                       const float* __restrict__ tem,
                       const int*   __restrict__ pos_flag,
                       bf16_t* __restrict__ attn_s) {
  int bh = blockIdx.x;
  int h  = bh % Hn;
  int w    = threadIdx.x >> 5;
  int lane = threadIdx.x & 31;
  int tm = w & 3;       // output tile row (d block)
  int tg = w >> 2;      // 0/1 -> col tiles {2tg, 2tg+1}

  const bf16_t* Q  = q_s + (size_t)bh * 64 * Nn;
  const bf16_t* Kp = k_s + (size_t)bh * 64 * Nn;

  v8f acc0 = (v8f)(0.0f), acc1 = (v8f)(0.0f);
  for (int k = 0; k < Nn; k += 32) {
    pf(Q  + (size_t)(tm * 16) * Nn + k + 128);
    pf(Kp + (size_t)(tg * 32) * Nn + k + 128);
    pf(Kp + (size_t)(tg * 32 + 16) * Nn + k + 128);
    v16bf a  = load_a_frag(Q + (size_t)(tm * 16) * Nn + k, Nn);
    v16bf b0 = load_b_frag(Kp + (size_t)(tg * 32) * Nn + k, Nn);
    v16bf b1 = load_b_frag(Kp + (size_t)(tg * 32 + 16) * Nn + k, Nn);
    acc0 = wmma_bf16(a, b0, acc0);
    acc1 = wmma_bf16(a, b1, acc1);
  }

  __shared__ float S[64][65];   // +1 pad
  int r0  = tm * 16 + ((lane >> 4) << 3);
  int c0a = tg * 32 + (lane & 15);
  int c0b = c0a + 16;
#pragma unroll
  for (int v = 0; v < 8; ++v) {
    S[r0 + v][c0a] = acc0[v];
    S[r0 + v][c0b] = acc1[v];
  }
  __syncthreads();

  int flag = *pos_flag;
  if (threadIdx.x < 64) {
    int r = threadIdx.x;
    float iq = invq[bh * 64 + r];
    float tp = temperature[h];
    const float* tm_row = tem + (size_t)h * 64 * 64 + (size_t)r * 64;
    float mx = -3.0e38f;
    for (int e = 0; e < 64; ++e) {
      float s = S[r][e] * iq * invk[bh * 64 + e] * tp;
      if (!flag) s += tm_row[e];
      S[r][e] = s;
      mx = fmaxf(mx, s);
    }
    float sum = 0.f;
    for (int e = 0; e < 64; ++e) {
      float ev = __expf(S[r][e] - mx);
      S[r][e] = ev;
      sum += ev;
    }
    float inv = 1.0f / sum;
    bf16_t* arow = attn_s + (size_t)bh * 4096 + (size_t)r * 64;
    for (int e0 = 0; e0 < 64; e0 += 8) {
      v8bf pkv;
#pragma unroll
      for (int t = 0; t < 8; ++t) pkv[t] = (bf16_t)(S[r][e0 + t] * inv);
      *(v8bf*)(arow + e0) = pkv;     // b128 stores, 128B/row contiguous
    }
  }
}

// ---------------------------------------------------------------------------
// Kernel 4: out^T[n,d] = sum_e v[e,n] * attn[d,e]  (per b,h; M=N, N'=64, K=64)
//   A = V^T:  v_s[(bh*4096+n)*64 + e]  (e-contiguous)
//   B = attn^T: lane col d, chunk contiguous e -> attn_s[d*64 + e]
// Result written as y[b,n, h*64+d] bf16 -> ready as A operand of projection.
// Block = 8 waves; wave w handles 16 rows of n x all 64 d.
// ---------------------------------------------------------------------------
__global__ void k_av(const bf16_t* __restrict__ v_s,
                     const bf16_t* __restrict__ attn_s,
                     bf16_t* __restrict__ y_s) {
  int bh = blockIdx.y;
  int b = bh / Hn, h = bh % Hn;
  int w    = threadIdx.x >> 5;
  int lane = threadIdx.x & 31;
  int n0 = blockIdx.x * 128 + w * 16;

  const bf16_t* V = v_s + (size_t)bh * Nn * 64;
  const bf16_t* A = attn_s + (size_t)bh * 4096;

  v8f acc[4];
#pragma unroll
  for (int j = 0; j < 4; ++j) acc[j] = (v8f)(0.0f);

#pragma unroll
  for (int k = 0; k < 64; k += 32) {
    v16bf a = load_a_frag(V + (size_t)n0 * 64 + k, 64);
#pragma unroll
    for (int j = 0; j < 4; ++j) {
      v16bf bf = load_b_frag(A + (size_t)(j * 16) * 64 + k, 64);
      acc[j] = wmma_bf16(a, bf, acc[j]);
    }
  }

#pragma unroll
  for (int j = 0; j < 4; ++j) {
    int d = j * 16 + (lane & 15);
#pragma unroll
    for (int v = 0; v < 8; ++v) {
      int n = n0 + ((lane >> 4) << 3) + v;
      y_s[((size_t)(b * Nn + n)) * Cn + h * 64 + d] = (bf16_t)acc[j][v];
    }
  }
}

// ---------------------------------------------------------------------------
// Kernel 5: projection GEMM. out[m,c] = sum_k y[m,k] * w_proj[c,k] + b_proj[c]
// M=32768, N'=768, K=768. Same tiling as kernel 1. f32 output.
// ---------------------------------------------------------------------------
__global__ void k_proj(const bf16_t* __restrict__ y_s,
                       const bf16_t* __restrict__ wpb,
                       const float* __restrict__ bias,
                       float* __restrict__ out) {
  int w    = threadIdx.x >> 5;
  int lane = threadIdx.x & 31;
  int wm = w & 1, wn = w >> 1;
  int m0 = blockIdx.y * 64 + wm * 32;
  int n0 = blockIdx.x * 256 + wn * 64;

  v8f acc[2][4];
#pragma unroll
  for (int i = 0; i < 2; ++i)
#pragma unroll
    for (int j = 0; j < 4; ++j) acc[i][j] = (v8f)(0.0f);

  for (int k = 0; k < Cn; k += 32) {
    pf(y_s + (size_t)m0 * Cn + k + 64);
    pf(y_s + (size_t)(m0 + 16) * Cn + k + 64);
    v16bf a0 = load_a_frag(y_s + (size_t)m0 * Cn + k, Cn);
    v16bf a1 = load_a_frag(y_s + (size_t)(m0 + 16) * Cn + k, Cn);
#pragma unroll
    for (int j = 0; j < 4; ++j) {
      v16bf bf = load_b_frag(wpb + (size_t)(n0 + j * 16) * Cn + k, Cn);
      acc[0][j] = wmma_bf16(a0, bf, acc[0][j]);
      acc[1][j] = wmma_bf16(a1, bf, acc[1][j]);
    }
  }

#pragma unroll
  for (int j = 0; j < 4; ++j) {
    int c = n0 + j * 16 + (lane & 15);
    float bv = bias[c];
#pragma unroll
    for (int i = 0; i < 2; ++i) {
      int mrow = m0 + i * 16 + ((lane >> 4) << 3);
#pragma unroll
      for (int v = 0; v < 8; ++v) {
        out[(size_t)(mrow + v) * Cn + c] = acc[i][j][v] + bv;
      }
    }
  }
}

// ---------------------------------------------------------------------------
// Launch
// ---------------------------------------------------------------------------
extern "C" void kernel_launch(void* const* d_in, const int* in_sizes, int n_in,
                              void* d_out, int out_size, void* d_ws, size_t ws_size,
                              hipStream_t stream) {
  (void)in_sizes; (void)n_in; (void)out_size; (void)ws_size;

  const float* x           = (const float*)d_in[0];   // [8,4096,768]
  const float* tem         = (const float*)d_in[1];   // [12,64,64]
  const float* temperature = (const float*)d_in[2];   // [12,1,1]
  const float* w_qkv       = (const float*)d_in[3];   // [2304,768]
  const float* w_proj      = (const float*)d_in[4];   // [768,768]
  const float* b_proj      = (const float*)d_in[5];   // [768]
  const int*   pos_flag    = (const int*)d_in[6];

  float* out = (float*)d_out;

  // Workspace layout (bytes, all 4KB-aligned offsets)
  char* ws = (char*)d_ws;
  bf16_t* x_bf   = (bf16_t*)(ws + 0);                  //  50331648
  bf16_t* wqkv_b = (bf16_t*)(ws + 50331648);           //   3538944
  bf16_t* wproj_b= (bf16_t*)(ws + 53870592);           //   1179648
  bf16_t* q_s    = (bf16_t*)(ws + 55050240);           //  50331648
  bf16_t* k_s    = (bf16_t*)(ws + 105381888);          //  50331648
  bf16_t* v_s    = (bf16_t*)(ws + 155713536);          //  50331648
  float*  invq   = (float*) (ws + 206045184);          //     24576
  float*  invk   = (float*) (ws + 206069760);          //     24576
  bf16_t* attn_s = (bf16_t*)(ws + 206094336);          //    786432
  bf16_t* y_s    = (bf16_t*)(ws + 206880768);          //  50331648
  // total ~257 MB

  // 0) converts (counts are /8, all tensors multiple of 8 elements)
  k_convert_bf16<<<4096, 256, 0, stream>>>(x, x_bf, (Mtot * Cn) / 8);
  k_convert_bf16<<<512, 256, 0, stream>>>(w_qkv, wqkv_b, (C3 * Cn) / 8);
  k_convert_bf16<<<256, 256, 0, stream>>>(w_proj, wproj_b, (Cn * Cn) / 8);

  // 1) QKV GEMM: grid (2304/256, 32768/64)
  k_gemm_qkv<<<dim3(C3 / 256, Mtot / 64), 256, 0, stream>>>(
      x_bf, wqkv_b, q_s, k_s, v_s);

  // 2) norms: one block per (bh,d) row
  k_norms<<<Bn * Hn * DHn, 256, 0, stream>>>(q_s, k_s, invq, invk);

  // 3) gram + softmax: one block per (b,h)
  k_attn<<<Bn * Hn, 256, 0, stream>>>(q_s, k_s, invq, invk,
                                      temperature, tem, pos_flag, attn_s);

  // 4) attn @ V (transposed): grid (4096/128, 96)
  k_av<<<dim3(Nn / 128, Bn * Hn), 256, 0, stream>>>(v_s, attn_s, y_s);

  // 5) projection: grid (768/256, 32768/64)
  k_proj<<<dim3(Cn / 256, Mtot / 64), 256, 0, stream>>>(y_s, wproj_b, b_proj, out);
}
